// RBlockEfficient_91225105367740
// MI455X (gfx1250) — matrix operand order repaired and checked by reading, not compiled, not run on hardware
//
#include <hip/hip_runtime.h>

typedef __attribute__((ext_vector_type(16))) __bf16 v16bf;
typedef __attribute__((ext_vector_type(8)))  __bf16 v8bf;
typedef __attribute__((ext_vector_type(4)))  __bf16 v4bf;
typedef __attribute__((ext_vector_type(2)))  __bf16 v2bf;
typedef __attribute__((ext_vector_type(8)))  float  v8f;
typedef __attribute__((ext_vector_type(4)))  float  v4f;

#define B_  16
#define C_  256
#define CI_ 128
#define W_  64
#define N_  4096
#define M_  1024

__device__ __forceinline__ v16bf mk16(v8bf a, v8bf b) {
  v16bf r;
#pragma unroll
  for (int j = 0; j < 8; ++j) { r[j] = a[j]; r[8 + j] = b[j]; }
  return r;
}
__device__ __forceinline__ v16bf mk16_4(v4bf a, v4bf b, v4bf c, v4bf d) {
  v16bf r;
#pragma unroll
  for (int j = 0; j < 4; ++j) { r[j] = a[j]; r[4+j] = b[j]; r[8+j] = c[j]; r[12+j] = d[j]; }
  return r;
}

// low 32 bits of a flat shared pointer = wave-relative LDS byte offset
__device__ __forceinline__ unsigned lds_off(const void* p) {
  return (unsigned)(unsigned long long)p;
}
// CDNA5 async DMA: global -> LDS, 16B per lane, tracked by ASYNCcnt
__device__ __forceinline__ void async_load_b128(unsigned ldsoff, const void* gptr) {
  asm volatile("global_load_async_to_lds_b128 %0, %1, off"
               :: "v"(ldsoff), "v"(gptr) : "memory");
}
__device__ __forceinline__ void wait_async0() {
  asm volatile("s_wait_asynccnt 0x0" ::: "memory");
}

// ---------------- K0: scalar precompute (u, alpha/beta, bf16 weights) ----------------
__global__ void k0_setup(const float* __restrict__ theta_w, const float* __restrict__ theta_b,
                         const float* __restrict__ phi_w,  const float* __restrict__ phi_b,
                         const float* __restrict__ psi_w,  const float* __restrict__ psi_b,
                         const float* __restrict__ rel_w0,
                         const float* __restrict__ W_w,    const float* __restrict__ W_b,
                         const float* __restrict__ bn_g,   const float* __restrict__ bn_b,
                         const float* __restrict__ bn_m,   const float* __restrict__ bn_v,
                         float* __restrict__ u, float* __restrict__ c0,
                         float* __restrict__ alpha, float* __restrict__ beta,
                         float* __restrict__ qkb,
                         __bf16* __restrict__ wqk, __bf16* __restrict__ wW) {
  const int t = threadIdx.x;
  const float invM = 1.0f / (float)M_;
  float acc = 0.f;
#pragma unroll 4
  for (int ci = 0; ci < CI_; ++ci) acc += rel_w0[ci] * theta_w[ci * C_ + t];
  u[t] = acc * invM;
  if (t == 0) {
    float c = 0.f;
    for (int ci = 0; ci < CI_; ++ci) c += rel_w0[ci] * theta_b[ci];
    c0[0] = c * invM;
  }
  float inv = bn_g[t] * rsqrtf(bn_v[t] + 1e-5f);
  alpha[t] = inv;
  beta[t]  = W_b[t] * inv + bn_b[t] - bn_m[t] * inv;
  qkb[t]   = (t < CI_) ? phi_b[t] : psi_b[t - CI_];
  for (int i = t; i < C_ * C_; i += 256) {
    int r = i >> 8, c = i & 255;
    float v = (r < CI_) ? phi_w[r * C_ + c] : psi_w[(r - CI_) * C_ + c];
    wqk[i] = (__bf16)v;
  }
  for (int i = t; i < C_ * CI_; i += 256) {
    int o = i >> 7, ci = i & 127;
    wW[i] = (__bf16)W_w[o * (CI_ + 1) + ci];  // drop all-zero structure column
  }
}

// ---------------- K1: phi/psi conv GEMM + 2x2 maxpool + s_t -------------------------
// grid (32,16): blockIdx.x = rp (pooled row). One WG computes all 256 output channels
// for one 128-column row pair, so x is read from HBM exactly once.
__global__ void __launch_bounds__(256) k1_convpool(
    const float* __restrict__ x, const float* __restrict__ u, const float* __restrict__ c0,
    const __bf16* __restrict__ wqk, const float* __restrict__ qkb,
    __bf16* __restrict__ phi_p, __bf16* __restrict__ psi_p, float* __restrict__ s_t) {
  __shared__ __bf16 ldsx[128 * 40];  // [col][c] transposed tile, padded stride 40
  const int tid = threadIdx.x;
  const int wave = tid >> 5, lane = tid & 31;
  const int rp = blockIdx.x;
  const int b  = blockIdx.y;
  const int nbase = rp * 2 * W_;                 // 128 cols: row 2rp (0..63), row 2rp+1 (64..127)
  const int q   = wave & 3;                      // 16-wide w quarter
  const int ch0 = (wave >> 2) * 128;             // wave's 128 output channels (8 subtiles)
  const int l15 = lane & 15;
  const int kb8  = (lane < 16) ? 0 : 8;
  const int kb16 = (lane < 16) ? 0 : 16;

  v8f acc[8][2];
#pragma unroll
  for (int t = 0; t < 8; ++t)
#pragma unroll
    for (int s = 0; s < 2; ++s) acc[t][s] = (v8f)(0.f);

  float stacc = 0.f;

  for (int kk = 0; kk < C_; kk += 32) {
    __syncthreads();
    // stage 32c x 128col of x, fp32 -> bf16, transposed: b128 global loads,
    // packed b32 LDS stores. 512 work items / 256 threads = 2 iters.
#pragma unroll
    for (int i = tid; i < 16 * 32; i += 256) {
      int cp = i >> 5;          // channel pair -> c = kk+2cp, kk+2cp+1
      int cq = i & 31;          // column quad  -> cols 4cq..4cq+3
      const v4f a0 = *(const v4f*)(x + ((size_t)b * C_ + kk + 2 * cp) * N_ + nbase + 4 * cq);
      const v4f a1 = *(const v4f*)(x + ((size_t)b * C_ + kk + 2 * cp + 1) * N_ + nbase + 4 * cq);
#pragma unroll
      for (int j = 0; j < 4; ++j) {
        v2bf p; p[0] = (__bf16)a0[j]; p[1] = (__bf16)a1[j];
        *(v2bf*)&ldsx[(4 * cq + j) * 40 + 2 * cp] = p;
      }
    }
    __syncthreads();
    if (tid < 128) {                            // fused s_t = u . x[:,n]
#pragma unroll
      for (int c = 0; c < 32; ++c)
        stacc += (float)ldsx[tid * 40 + c] * u[kk + c];
    }
    v16bf bf[2];
#pragma unroll
    for (int s = 0; s < 2; ++s) {               // s=0 -> row 2rp, s=1 -> row 2rp+1
      int col = q * 16 + l15 + s * 64;
      const v4bf* p = (const v4bf*)&ldsx[col * 40 + kb8];
      const v4bf* p2 = (const v4bf*)&ldsx[col * 40 + 16 + kb8];
      bf[s] = mk16_4(p[0], p[1], p2[0], p2[1]);
    }
#pragma unroll
    for (int t = 0; t < 8; ++t) {
      int row = ch0 + t * 16 + l15;
      const v8bf* a0 = (const v8bf*)(wqk + row * C_ + kk + kb8);
      const v8bf* a1 = (const v8bf*)(wqk + row * C_ + kk + 16 + kb8);
      v16bf af = mk16(a0[0], a1[0]);
#pragma unroll
      for (int s = 0; s < 2; ++s)
        acc[t][s] = __builtin_amdgcn_wmma_f32_16x16x32_bf16(
            false, af, false, bf[s], (short)0, acc[t][s], false, false);
    }
  }

  // 2x2 maxpool: vertical max across the two accumulators, horizontal via lane-pair shuffle
  const int rh = (lane < 16) ? 0 : 8;
#pragma unroll
  for (int t = 0; t < 8; ++t) {
#pragma unroll
    for (int r = 0; r < 8; ++r) {
      int ch = ch0 + t * 16 + r + rh;
      float m0 = fmaxf(acc[t][0][r], acc[t][1][r]);
      float m1 = __shfl_xor(m0, 1, 32);
      if ((lane & 1) == 0) {
        float pv = fmaxf(m0, m1) + qkb[ch];
        int m = rp * 32 + q * 8 + (l15 >> 1);
        if (ch < CI_)
          phi_p[((size_t)b * CI_ + ch) * M_ + m] = (__bf16)pv;
        else
          psi_p[((size_t)b * CI_ + (ch - CI_)) * M_ + m] = (__bf16)pv;
      }
    }
  }
  if (tid < 128)
    s_t[(size_t)b * N_ + nbase + tid] = stacc + c0[0];
}

// ---------------- K2: s_p = (1/M) rel_w1 . phi_pooled --------------------------------
__global__ void k2_sp(const __bf16* __restrict__ phi_p, const float* __restrict__ rel_w1,
                      float* __restrict__ s_p) {
  const int b = blockIdx.y;
  const int m = blockIdx.x * 256 + threadIdx.x;
  float acc = 0.f;
#pragma unroll 8
  for (int ci = 0; ci < CI_; ++ci)
    acc += rel_w1[ci] * (float)phi_p[((size_t)b * CI_ + ci) * M_ + m];
  s_p[(size_t)b * M_ + m] = acc * (1.0f / (float)M_);
}

// ---------------- K3: y = relu(s_t + s_p) @ psi^T, A generated in registers ----------
// grid (32,16): 128 n-rows per WG; wave = 32n x 64ci
__global__ void __launch_bounds__(256) k3_attn(
    const float* __restrict__ s_t, const float* __restrict__ s_p,
    const __bf16* __restrict__ psi_p, __bf16* __restrict__ y) {
  __shared__ float ldsp[M_];
  const int tid = threadIdx.x, wave = tid >> 5, lane = tid & 31;
  const int b  = blockIdx.y;
  const int nb = blockIdx.x * 128;
  const int ng = wave & 3;              // two n-subtiles: 2ng, 2ng+1
  const int ch = (wave >> 2) * 64;      // ci base
  const int l15 = lane & 15;
  const int kb8  = (lane < 16) ? 0 : 8;
  const int kb16 = (lane < 16) ? 0 : 16;

  // async DMA the 4KB s_p row into LDS: 256 lanes x b128, one issue
  async_load_b128(lds_off(&ldsp[tid * 4]), s_p + (size_t)b * M_ + tid * 4);
  wait_async0();
  __syncthreads();

  float stv[2];
#pragma unroll
  for (int np = 0; np < 2; ++np)
    stv[np] = s_t[(size_t)b * N_ + nb + (2 * ng + np) * 16 + l15];

  v8f acc[2][4];
#pragma unroll
  for (int np = 0; np < 2; ++np)
#pragma unroll
    for (int c = 0; c < 4; ++c) acc[np][c] = (v8f)(0.f);

  for (int mm = 0; mm < M_; mm += 32) {
    v16bf bf[4];
#pragma unroll
    for (int c = 0; c < 4; ++c) {
      int ci = ch + c * 16 + l15;
      const __bf16* base = psi_p + ((size_t)b * CI_ + ci) * M_ + mm + kb16;
      const v8bf* p = (const v8bf*)base;
      bf[c] = mk16(p[0], p[1]);
      if (mm + 32 < M_) __builtin_prefetch(base + 32, 0, 1);  // global_prefetch next K-slice
    }
#pragma unroll
    for (int np = 0; np < 2; ++np) {
      v16bf af;                          // f tile generated on the fly (no HBM traffic)
#pragma unroll
      for (int j = 0; j < 16; ++j) {
        int ml = mm + kb16 + j;
        af[j] = (__bf16)fmaxf(stv[np] + ldsp[ml], 0.f);
      }
#pragma unroll
      for (int c = 0; c < 4; ++c)
        acc[np][c] = __builtin_amdgcn_wmma_f32_16x16x32_bf16(
            false, af, false, bf[c], (short)0, acc[np][c], false, false);
    }
  }
  const int rh = (lane < 16) ? 0 : 8;
#pragma unroll
  for (int np = 0; np < 2; ++np)
#pragma unroll
    for (int c = 0; c < 4; ++c)
#pragma unroll
      for (int r = 0; r < 8; ++r) {
        int n  = nb + (2 * ng + np) * 16 + r + rh;
        int ci = ch + c * 16 + l15;
        y[((size_t)b * N_ + n) * CI_ + ci] = (__bf16)acc[np][c][r];
      }
}

// ---------------- K4: out = alpha*(W_w @ y) + beta + x (fused BN + residual) ---------
// grid (64,16): blockIdx.x = og + 2*ngrp. The y K-slice (128n x 32ci, 8KB) is
// async-DMA'd to LDS once per K-step so the 4 o-subtile waves share one L2 fetch.
__global__ void __launch_bounds__(256) k4_out(
    const __bf16* __restrict__ y, const __bf16* __restrict__ wW,
    const float* __restrict__ alpha, const float* __restrict__ beta,
    const float* __restrict__ x, float* __restrict__ out) {
  __shared__ __bf16 ldsy[128 * 32];     // [n_local][ci32]
  const int tid = threadIdx.x, wave = tid >> 5, lane = tid & 31;
  const int b  = blockIdx.y;
  const int og = (blockIdx.x & 1) * 128;
  const int nb = (blockIdx.x >> 1) * 128;
  const int op2 = (wave & 3) * 2;       // wave's two o-subtiles
  const int nh  = (wave >> 2) * 64;     // wave's 64-wide n range
  const int l15 = lane & 15;
  const int kb8  = (lane < 16) ? 0 : 8;
  const int kb16 = (lane < 16) ? 0 : 16;

  v8f acc[2][4];
#pragma unroll
  for (int op = 0; op < 2; ++op)
#pragma unroll
    for (int c = 0; c < 4; ++c) acc[op][c] = (v8f)(0.f);

#pragma unroll
  for (int kk = 0; kk < CI_; kk += 32) {
    __syncthreads();
    // 128 rows x 64B = 8KB -> 512 b128 chunks, 2 per thread, via async DMA
#pragma unroll
    for (int it = 0; it < 2; ++it) {
      int chunk = tid + it * 256;
      int row = chunk >> 2, qtr = chunk & 3;
      async_load_b128(lds_off(&ldsy[row * 32 + qtr * 8]),
                      y + ((size_t)b * N_ + nb + row) * CI_ + kk + qtr * 8);
    }
    wait_async0();
    __syncthreads();

    v16bf af[2];
#pragma unroll
    for (int op = 0; op < 2; ++op) {
      int o = og + (op2 + op) * 16 + l15;
      const v8bf* a0 = (const v8bf*)(wW + o * CI_ + kk + kb8);
      const v8bf* a1 = (const v8bf*)(wW + o * CI_ + kk + 16 + kb8);
      af[op] = mk16(a0[0], a1[0]);
    }
    v16bf bf[4];
#pragma unroll
    for (int c = 0; c < 4; ++c) {
      int nl = nh + c * 16 + l15;
      const v8bf* p = (const v8bf*)&ldsy[nl * 32 + kb8];
      const v8bf* p2 = (const v8bf*)&ldsy[nl * 32 + 16 + kb8];
      bf[c] = mk16(p[0], p2[0]);
    }
#pragma unroll
    for (int op = 0; op < 2; ++op)
#pragma unroll
      for (int c = 0; c < 4; ++c)
        acc[op][c] = __builtin_amdgcn_wmma_f32_16x16x32_bf16(
            false, af[op], false, bf[c], (short)0, acc[op][c], false, false);
  }

  const int rh = (lane < 16) ? 0 : 8;
#pragma unroll
  for (int op = 0; op < 2; ++op)
#pragma unroll
    for (int c = 0; c < 4; ++c)
#pragma unroll
      for (int r = 0; r < 8; ++r) {
        int o = og + (op2 + op) * 16 + r + rh;
        int n = nb + nh + c * 16 + l15;
        size_t xi = ((size_t)b * C_ + o) * N_ + n;
        out[xi] = alpha[o] * acc[op][c][r] + beta[o] + x[xi];
      }
}

// ---------------- host launcher ------------------------------------------------------
extern "C" void kernel_launch(void* const* d_in, const int* in_sizes, int n_in,
                              void* d_out, int out_size, void* d_ws, size_t ws_size,
                              hipStream_t stream) {
  const float* x       = (const float*)d_in[0];
  const float* theta_w = (const float*)d_in[1];
  const float* theta_b = (const float*)d_in[2];
  const float* phi_w   = (const float*)d_in[3];
  const float* phi_b   = (const float*)d_in[4];
  const float* psi_w   = (const float*)d_in[5];
  const float* psi_b   = (const float*)d_in[6];
  const float* rel_w0  = (const float*)d_in[7];
  const float* rel_w1  = (const float*)d_in[8];
  const float* W_w     = (const float*)d_in[9];
  const float* W_b     = (const float*)d_in[10];
  const float* bn_g    = (const float*)d_in[11];
  const float* bn_b    = (const float*)d_in[12];
  const float* bn_m    = (const float*)d_in[13];
  const float* bn_v    = (const float*)d_in[14];

  char* ws = (char*)d_ws;
  float*  u     = (float*)(ws + 0);          //  256 f
  float*  c0    = (float*)(ws + 1024);       //  1 f
  float*  alpha = (float*)(ws + 1280);       //  256 f
  float*  beta  = (float*)(ws + 2304);       //  256 f
  float*  qkb   = (float*)(ws + 3328);       //  256 f
  float*  s_t   = (float*)(ws + 4608);       //  B*N f
  float*  s_p   = (float*)(ws + 266752);     //  B*M f
  __bf16* wqk   = (__bf16*)(ws + 332288);    //  256*256 bf16
  __bf16* wW    = (__bf16*)(ws + 463360);    //  256*128 bf16
  __bf16* phi_p = (__bf16*)(ws + 528896);    //  B*CI*M bf16
  __bf16* psi_p = (__bf16*)(ws + 4723200);   //  B*CI*M bf16
  __bf16* ybf   = (__bf16*)(ws + 8917504);   //  B*N*CI bf16 (ends at ~25.7 MB)
  float*  out   = (float*)d_out;

  k0_setup<<<1, 256, 0, stream>>>(theta_w, theta_b, phi_w, phi_b, psi_w, psi_b,
                                  rel_w0, W_w, W_b, bn_g, bn_b, bn_m, bn_v,
                                  u, c0, alpha, beta, qkb, wqk, wW);
  k1_convpool<<<dim3(32, B_), 256, 0, stream>>>(x, u, c0, wqk, qkb, phi_p, psi_p, s_t);
  k2_sp<<<dim3(M_ / 256, B_), 256, 0, stream>>>(phi_p, rel_w1, s_p);
  k3_attn<<<dim3(N_ / 128, B_), 256, 0, stream>>>(s_t, s_p, psi_p, ybf);
  k4_out<<<dim3(64, B_), 256, 0, stream>>>(ybf, wW, alpha, beta, x, out);
}
